// MultiHeadAttention_773094114093
// MI455X (gfx1250) — matrix operand order repaired and checked by reading, not compiled, not run on hardware
//
#include <hip/hip_runtime.h>
#include <cstdint>

typedef __bf16 bf16;
typedef __attribute__((ext_vector_type(16))) __bf16 v16bf;
typedef __attribute__((ext_vector_type(8)))  __bf16 v8bf;
typedef __attribute__((ext_vector_type(4)))  __bf16 v4bf;
typedef __attribute__((ext_vector_type(8)))  float  v8f;
typedef __attribute__((ext_vector_type(4)))  unsigned int u32x4;
typedef __attribute__((ext_vector_type(8)))  int  i32x8;
typedef __attribute__((ext_vector_type(4)))  int  i32x4;

#define B_  4
#define N_  2048
#define E_  768
#define H_  8
#define D_  96
#define SCALING 0.10206207261596575f  // 96^-0.5

__device__ __forceinline__ v16bf pack16(v8bf lo, v8bf hx) {
    return __builtin_shufflevector(lo, hx, 0,1,2,3,4,5,6,7,8,9,10,11,12,13,14,15);
}
__device__ __forceinline__ v16bf loadpair8(const bf16* p) {   // p, p+16
    v8bf lo = *(const v8bf*)p;
    v8bf hx = *(const v8bf*)(p + 16);
    return pack16(lo, hx);
}
__device__ __forceinline__ v8f wmma_bf16(v16bf a, v16bf b, v8f c) {
    return __builtin_amdgcn_wmma_f32_16x16x32_bf16(false, a, false, b, (short)0, c, false, false);
}
__device__ __forceinline__ float halfmax(float x) {   // reduce over 16-lane half-wave
    #pragma unroll
    for (int m = 1; m < 16; m <<= 1) x = fmaxf(x, __shfl_xor(x, m, 32));
    return x;
}
__device__ __forceinline__ float halfsum(float x) {
    #pragma unroll
    for (int m = 1; m < 16; m <<= 1) x += __shfl_xor(x, m, 32);
    return x;
}

// ---------------------------------------------------------------------------
// TDM: issue a 2D tensor_load_to_lds (bf16 elements), LDS padding enabled:
// 4 pad DWORDs (16B) inserted after every 16 stored DWORDs (64B) -> a 32-elem
// chunk occupies 40 bf16 of LDS, de-conflicting banks for fragment reads.
// D# packing per CDNA5 ISA ch.8 (count=1, type=2 "image", data_size=2B).
// This toolchain exposes the 6-arg builtin: (g0, g1, g2, g3, g4, cpol).
// ---------------------------------------------------------------------------
__device__ __forceinline__ void tdm_load_2d(unsigned lds_off, const bf16* gptr,
                                            unsigned td0, unsigned td1,
                                            unsigned tile0, unsigned tile1,
                                            unsigned long long stride0)
{
    unsigned long long ga = (unsigned long long)(uintptr_t)gptr;
    u32x4 g0;
    g0[0] = 1u;                                             // count=1, user mode
    g0[1] = lds_off;                                        // lds_addr
    g0[2] = (unsigned)ga;                                   // global_addr[31:0]
    g0[3] = (unsigned)((ga >> 32) & 0x01FFFFFFu) | 0x80000000u; // ga[56:32] | type=2
    i32x8 g1 = {};
    // data_size=1(2B) @16 | pad_enable @20 | pad_interval=3(16 dw) @22 | pad_amount=3(4 dw) @25
    g1[0] = (int)((1u << 16) | (1u << 20) | (3u << 22) | (3u << 25));
    g1[1] = (int)((td0 & 0xFFFFu) << 16);                   // tensor_dim0[15:0]
    g1[2] = (int)(((td0 >> 16) & 0xFFFFu) | ((td1 & 0xFFFFu) << 16));
    g1[3] = (int)(((td1 >> 16) & 0xFFFFu) | ((tile0 & 0xFFFFu) << 16));
    g1[4] = (int)(tile1 & 0xFFFFu);                         // tile_dim1 (tile_dim2=0)
    g1[5] = (int)(stride0 & 0xFFFFFFFFull);                 // tensor_dim0_stride[31:0]
    g1[6] = (int)((stride0 >> 32) & 0xFFFFull);             // stride0[47:32]
    g1[7] = 0;
    i32x4 z4 = {};
    i32x8 z8 = {};
    __builtin_amdgcn_tensor_load_to_lds(g0, g1, z4, z4, z8, 0);
}

// ---------------------------------------------------------------------------
// Tiled GEMM (128x128x32), bf16 WMMA, f32 accumulate.
//   AF32=true : A is f32, converted to bf16 while staging (QKV projection)
//   EPI=1    : qkv epilogue -> scatter Q[bh][n][d], K[bh][n][d], Vt[bh][d][n] (bf16)
//   EPI=0    : C = A*W + bias, f32 row-major out (output projection)
// ---------------------------------------------------------------------------
template<bool AF32, int EPI>
__global__ __launch_bounds__(256) void gemm_bf16(
    const float* __restrict__ Af, const bf16* __restrict__ Ab,
    const float* __restrict__ W,  const float* __restrict__ bias,
    float* __restrict__ Cout,
    bf16* __restrict__ Qo, bf16* __restrict__ Ko, bf16* __restrict__ Vo,
    int M, int Nn, int K)
{
    const int tid  = threadIdx.x;
    const int wave = tid >> 5, lane = tid & 31, ln = lane & 15, hi = lane >> 4;
    const int wm = wave >> 1;          // 0..3 : 32-row strip
    const int wn = wave & 1;           // 0..1 : 64-col strip
    const int m0 = blockIdx.y * 128;
    const int n0 = blockIdx.x * 128;

    __shared__ bf16 ldsA[128][40];     // [row][k]  (pad 32->40, 80B rows)
    __shared__ bf16 ldsBt[128][40];    // [col][k]  transposed B tile

    v8f acc[2][4] = {};

    for (int k0 = 0; k0 < K; k0 += 32) {
        __syncthreads();
        if (AF32) {
            #pragma unroll
            for (int i = tid; i < 1024; i += 256) {        // 128x32 f32, float4 chunks
                int row = i >> 3, c4 = (i & 7) * 4;
                const float4 v = *(const float4*)&Af[(size_t)(m0 + row) * K + k0 + c4];
                v4bf w; w[0] = (bf16)v.x; w[1] = (bf16)v.y; w[2] = (bf16)v.z; w[3] = (bf16)v.w;
                *(v4bf*)&ldsA[row][c4] = w;
            }
        } else {
            #pragma unroll
            for (int i = tid; i < 512; i += 256) {         // 128x32 bf16, v8bf chunks
                int row = i >> 2, cj = (i & 3) * 8;
                *(v8bf*)&ldsA[row][cj] = *(const v8bf*)&Ab[(size_t)(m0 + row) * K + k0 + cj];
            }
        }
        #pragma unroll
        for (int i = tid; i < 1024; i += 256) {            // 32x128 f32 -> transposed bf16
            int kr = i >> 5, nc = (i & 31) * 4;
            const float4 v = *(const float4*)&W[(size_t)(k0 + kr) * Nn + n0 + nc];
            ldsBt[nc + 0][kr] = (bf16)v.x; ldsBt[nc + 1][kr] = (bf16)v.y;
            ldsBt[nc + 2][kr] = (bf16)v.z; ldsBt[nc + 3][kr] = (bf16)v.w;
        }
        __syncthreads();

        v16bf afr[2];
        #pragma unroll
        for (int i = 0; i < 2; ++i) {
            const bf16* p = &ldsA[wm * 32 + i * 16 + ln][hi * 8];
            v8bf lo = *(const v8bf*)p;
            v8bf hx = *(const v8bf*)(p + 16);
            afr[i] = pack16(lo, hx);
        }
        #pragma unroll
        for (int j = 0; j < 4; ++j) {
            v16bf bfr = loadpair8(&ldsBt[wn * 64 + j * 16 + ln][hi * 8]);
            #pragma unroll
            for (int i = 0; i < 2; ++i)
                acc[i][j] = wmma_bf16(afr[i], bfr, acc[i][j]);
        }
    }

    // Epilogue. C layout: VGPR r -> row r + 8*hi, col = ln.
    #pragma unroll
    for (int i = 0; i < 2; ++i)
    #pragma unroll
    for (int j = 0; j < 4; ++j)
    #pragma unroll
    for (int r = 0; r < 8; ++r) {
        int row = m0 + wm * 32 + i * 16 + r + hi * 8;
        int col = n0 + wn * 64 + j * 16 + ln;
        float val = acc[i][j][r] + bias[col];
        if (EPI == 0) {
            Cout[(size_t)row * Nn + col] = val;
        } else {
            // reference packs last axis as (h, d, qkv): col = h*288 + d*3 + which
            int b = row >> 11, nseq = row & 2047;
            int h = col / (D_ * 3);
            int d = (col / 3) % D_;
            int which = col % 3;
            bf16 bv = (bf16)val;
            size_t bh = (size_t)b * H_ + h;
            if (which == 0)      Qo[(bh * N_ + nseq) * D_ + d] = bv;
            else if (which == 1) Ko[(bh * N_ + nseq) * D_ + d] = bv;
            else                 Vo[(bh * D_ + d) * N_ + nseq] = bv;   // V transposed
        }
    }
}

// ---------------------------------------------------------------------------
// Flash attention: 8 waves x 16 queries per block; 64-key tiles double-
// buffered in LDS via the Tensor Data Mover (wave 0 issues, TENSORcnt gates);
// online softmax; bf16 WMMA for QK^T and PV.
// LDS tile geometry (TDM pad: +8 bf16 after every 32 stored bf16):
//   K tile : 64 keys x 96 d  -> row stride 120 bf16, chunk(c) at c*40
//   V tile : 96 d  x 64 keys -> row stride  80 bf16, chunk(c) at c*40
// ---------------------------------------------------------------------------
__global__ __launch_bounds__(256) void attn_fa(
    const bf16* __restrict__ Qg, const bf16* __restrict__ Kg,
    const bf16* __restrict__ Vg, bf16* __restrict__ Og)
{
    const int bh = blockIdx.y;                 // b*H + h
    const int q0 = blockIdx.x * 128;
    const int tid = threadIdx.x;
    const int wave = tid >> 5, lane = tid & 31, ln = lane & 15, hi = lane >> 4;
    const int qw = q0 + wave * 16;

    const bf16* Qb = Qg + (size_t)bh * N_ * D_;
    const bf16* Kb = Kg + (size_t)bh * N_ * D_;
    const bf16* Vb = Vg + (size_t)bh * D_ * N_;   // [d][n]

    __shared__ bf16 ldsK[2][64 * 120];
    __shared__ bf16 ldsV[2][96 * 80];
    __shared__ bf16 ldsP[8][16][72];   // per-wave P tile, 144B rows

    // Q fragments, resident for whole block: 3 chunks of d (A-layout)
    v16bf qa[3];
    #pragma unroll
    for (int c = 0; c < 3; ++c) {
        const bf16* p = Qb + (size_t)(qw + ln) * D_ + c * 32 + hi * 8;
        v8bf lo = *(const v8bf*)p;
        v8bf hx = *(const v8bf*)(p + 16);
        qa[c] = pack16(lo, hx);
    }

    v8f o[6] = {};
    float m[8], l[8];
    #pragma unroll
    for (int r = 0; r < 8; ++r) { m[r] = -1e30f; l[r] = 0.0f; }

    const int NT = N_ / 64;   // 32 key tiles

    if (wave == 0) {          // prologue: DMA tile 0 into buffer 0
        tdm_load_2d((unsigned)(uintptr_t)&ldsK[0][0], Kb, D_, N_, D_, 64, (unsigned long long)D_);
        tdm_load_2d((unsigned)(uintptr_t)&ldsV[0][0], Vb, N_, D_, 64, D_, (unsigned long long)N_);
    }

    for (int kt = 0; kt < NT; ++kt) {
        const int buf = kt & 1;
        if (wave == 0) {
            if (kt + 1 < NT) {   // prefetch next tile into other buffer
                tdm_load_2d((unsigned)(uintptr_t)&ldsK[buf ^ 1][0],
                            Kb + (size_t)(kt + 1) * 64 * D_, D_, N_, D_, 64,
                            (unsigned long long)D_);
                tdm_load_2d((unsigned)(uintptr_t)&ldsV[buf ^ 1][0],
                            Vb + (size_t)(kt + 1) * 64, N_, D_, 64, D_,
                            (unsigned long long)N_);
                __builtin_amdgcn_s_wait_tensorcnt((short)2);  // current tile done
            } else {
                __builtin_amdgcn_s_wait_tensorcnt((short)0);
            }
        }
        __syncthreads();
        const bf16* bk = &ldsK[buf][0];
        const bf16* bv = &ldsV[buf][0];

        // S = Q K^T for 64 keys -> four 16x16 C fragments
        v8f s[4] = {};
        #pragma unroll
        for (int c = 0; c < 3; ++c) {
            #pragma unroll
            for (int j = 0; j < 4; ++j) {
                v16bf bfr = loadpair8(bk + (j * 16 + ln) * 120 + c * 40 + hi * 8);
                s[j] = wmma_bf16(qa[c], bfr, s[j]);
            }
        }

        // online softmax per row (row = r + 8*hi lives in a 16-lane half-wave)
        #pragma unroll
        for (int r = 0; r < 8; ++r) {
            float tmax = fmaxf(fmaxf(s[0][r], s[1][r]), fmaxf(s[2][r], s[3][r]));
            tmax = halfmax(tmax);
            float mn = fmaxf(m[r], tmax);
            float alpha = __expf(m[r] - mn);
            m[r] = mn;
            float p0 = __expf(s[0][r] - mn), p1 = __expf(s[1][r] - mn);
            float p2 = __expf(s[2][r] - mn), p3 = __expf(s[3][r] - mn);
            l[r] = l[r] * alpha + halfsum(p0 + p1 + p2 + p3);
            #pragma unroll
            for (int dt = 0; dt < 6; ++dt) o[dt][r] *= alpha;
            bf16* pp = &ldsP[wave][r + 8 * hi][0];
            pp[ln] = (bf16)p0; pp[16 + ln] = (bf16)p1;
            pp[32 + ln] = (bf16)p2; pp[48 + ln] = (bf16)p3;
        }
        // same-wave LDS ordering: stores retire before the P fragment loads
        asm volatile("s_wait_dscnt 0" ::: "memory");

        // P (C-layout) -> A-layout via per-wave LDS, then O += P V
        #pragma unroll
        for (int kb = 0; kb < 2; ++kb) {
            const bf16* pr = &ldsP[wave][ln][kb * 32];
            v16bf pa = pack16(*(const v8bf*)&pr[hi * 8], *(const v8bf*)&pr[16 + hi * 8]);
            #pragma unroll
            for (int dt = 0; dt < 6; ++dt) {
                v16bf vb = loadpair8(bv + (dt * 16 + ln) * 80 + kb * 40 + hi * 8);
                o[dt] = wmma_bf16(pa, vb, o[dt]);
            }
        }
        __syncthreads();   // all waves done with this buffer
    }

    // normalize (reference: softmax then * SCALING), store bf16 [b][n][h*D+d]
    const int b = bh >> 3, h = bh & 7;
    #pragma unroll
    for (int r = 0; r < 8; ++r) {
        float inv = SCALING / l[r];
        int qrow = qw + r + hi * 8;
        #pragma unroll
        for (int dt = 0; dt < 6; ++dt) {
            float val = o[dt][r] * inv;
            Og[((size_t)b * N_ + qrow) * E_ + h * D_ + dt * 16 + ln] = (bf16)val;
        }
    }
}

extern "C" void kernel_launch(void* const* d_in, const int* in_sizes, int n_in,
                              void* d_out, int out_size, void* d_ws, size_t ws_size,
                              hipStream_t stream) {
    const float* x     = (const float*)d_in[0];   // [B,N,E]
    const float* Wqkv  = (const float*)d_in[1];   // [E,3E]
    const float* bqkv  = (const float*)d_in[2];   // [3E]
    const float* Wproj = (const float*)d_in[3];   // [E,E]
    const float* bproj = (const float*)d_in[4];   // [E]
    float* out = (float*)d_out;

    const size_t nQ = (size_t)B_ * H_ * N_ * D_;  // == B*N*E
    bf16* Qb = (bf16*)d_ws;
    bf16* Kb = Qb + nQ;
    bf16* Vt = Kb + nQ;                            // transposed V [bh][d][n]
    bf16* AO = Vt + nQ;                            // attention out bf16 [B*N][E]

    // 1) fused QKV projection + bias + (h,d,qkv) de-interleave scatter
    gemm_bf16<true, 1><<<dim3(3 * E_ / 128, B_ * N_ / 128), 256, 0, stream>>>(
        x, nullptr, Wqkv, bqkv, nullptr, Qb, Kb, Vt, B_ * N_, 3 * E_, E_);

    // 2) flash attention (TDM-staged K/V tiles, online softmax)
    attn_fa<<<dim3(N_ / 128, B_ * H_), 256, 0, stream>>>(Qb, Kb, Vt, AO);

    // 3) output projection + bias, f32 out
    gemm_bf16<false, 0><<<dim3(E_ / 128, B_ * N_ / 128), 256, 0, stream>>>(
        nullptr, AO, Wproj, bproj, out, nullptr, nullptr, nullptr, B_ * N_, E_, E_);
}
